// CovCNNandRNN_28595892257505
// MI455X (gfx1250) — matrix operand order repaired and checked by reading, not compiled, not run on hardware
//
#include <hip/hip_runtime.h>
#include <hip/hip_bf16.h>
#include <math.h>

#define EPSV 1e-5f
#define PI_F 3.14159265358979323846f

typedef __attribute__((ext_vector_type(16))) __bf16 v16bf;
typedef __attribute__((ext_vector_type(8)))  __bf16 bf16x8;
typedef __attribute__((ext_vector_type(8)))  float  v8f;

__device__ __forceinline__ unsigned short f2bf(float f) {
  unsigned int u = __float_as_uint(f);
  u += 0x7FFFu + ((u >> 16) & 1u);            // round-to-nearest-even
  return (unsigned short)(u >> 16);
}
__device__ __forceinline__ float sigm(float x) { return 1.0f / (1.0f + __expf(-x)); }

// 32-bit LDS byte address of a generic pointer into __shared__ (for async-to-LDS dst)
__device__ __forceinline__ unsigned lds_addr_of(const void* p) {
  return (unsigned)(unsigned long long)(const __attribute__((address_space(3))) void*)p;
}

// GLOBAL_LOAD_ASYNC_TO_LDS_B128: per-lane 16-byte copy from global memory into LDS; increments ASYNCcnt
__device__ __forceinline__ void async_load_b128(unsigned lds_addr, const void* gaddr) {
  asm volatile("global_load_async_to_lds_b128 %0, %1, off"
               :
               : "v"(lds_addr), "v"((unsigned long long)(size_t)gaddr)
               : "memory");
}

template <int N>
__device__ __forceinline__ void wait_asynccnt() {
#if __has_builtin(__builtin_amdgcn_s_wait_asynccnt)
  __builtin_amdgcn_s_wait_asynccnt((short)N);
#else
  asm volatile("s_wait_asynccnt %0" : : "i"(N) : "memory");
#endif
}

// ---------------------------------------------------------------------------
// K1: per-batch analytic BN2d stats of cov = X X^T, without forming cov.
//   sum(cov)   = sum_t colsum[t]^2
//   sum(cov^2) = || X^T X ||_F^2  (40x40 Gram)
// ---------------------------------------------------------------------------
__global__ __launch_bounds__(256) void stats_kernel(const float* __restrict__ x,
                                                    float* __restrict__ stats) {
  const int b = blockIdx.x;
  const int tid = threadIdx.x;
  __shared__ float lx[128 * 40];
  __shared__ float red[256];
  float acc[7];
#pragma unroll
  for (int i = 0; i < 7; ++i) acc[i] = 0.f;
  float colsum = 0.f;
  const float* xb = x + (size_t)b * 2025 * 40;
  for (int r0 = 0; r0 < 2025; r0 += 128) {
    for (int e = tid; e < 128 * 40; e += 256) {
      int r = e / 40, t = e - r * 40;
      int rg = r0 + r;
      lx[e] = (rg < 2025) ? xb[(size_t)rg * 40 + t] : 0.f;
    }
    __syncthreads();
#pragma unroll
    for (int i = 0; i < 7; ++i) {
      int p = tid + i * 256;
      if (p < 1600) {
        int t1 = p / 40, t2 = p - t1 * 40;
        float s = 0.f;
        for (int r = 0; r < 128; ++r) s += lx[r * 40 + t1] * lx[r * 40 + t2];
        acc[i] += s;
      }
    }
    if (tid < 40) {
      float s = 0.f;
      for (int r = 0; r < 128; ++r) s += lx[r * 40 + tid];
      colsum += s;
    }
    __syncthreads();
  }
  float sumsq = 0.f;
#pragma unroll
  for (int i = 0; i < 7; ++i)
    if (tid + i * 256 < 1600) sumsq += acc[i] * acc[i];
  float sumcov = (tid < 40) ? colsum * colsum : 0.f;

  red[tid] = sumcov; __syncthreads();
  for (int s = 128; s > 0; s >>= 1) { if (tid < s) red[tid] += red[tid + s]; __syncthreads(); }
  float totc = red[0];
  __syncthreads();
  red[tid] = sumsq; __syncthreads();
  for (int s = 128; s > 0; s >>= 1) { if (tid < s) red[tid] += red[tid + s]; __syncthreads(); }
  if (tid == 0) { stats[b] = totc; stats[32 + b] = red[0]; }
}

// K1b: fold BN2d into conv1 affine:  y = inv * (w (*) cov) + beta_eff
__global__ void finalize1_kernel(const float* __restrict__ stats, const float* __restrict__ w1,
                                 const float* __restrict__ conv1_b, const float* __restrict__ g,
                                 const float* __restrict__ bb, float* __restrict__ scal) {
  if (threadIdx.x == 0) {
    float S1 = 0.f, S2 = 0.f;
    for (int i = 0; i < 32; ++i) { S1 += stats[i]; S2 += stats[32 + i]; }
    const float N = 32.f * 2025.f * 2025.f;
    float mean = S1 / N;
    float var  = S2 / N - mean * mean;
    float inv  = g[0] * rsqrtf(var + EPSV);
    float sw = 0.f;
    for (int i = 0; i < 81; ++i) sw += w1[i];
    scal[0] = inv;
    scal[1] = (bb[0] - mean * inv) * sw + conv1_b[0];
  }
}

// ---------------------------------------------------------------------------
// K2: pack bf16 GEMM operands.  K = 360 (j,t), padded to 384.
//   A'[b][oy][j*40+t] = sum_i w1[i][j] * x[b][4oy+i][t]
//   D [b][ox][j*40+t] = x[b][4ox+j][t]
// ---------------------------------------------------------------------------
__global__ void pack_kernel(const float* __restrict__ x, const float* __restrict__ w1,
                            unsigned short* __restrict__ A, unsigned short* __restrict__ D) {
  long long idx = (long long)blockIdx.x * blockDim.x + threadIdx.x;
  const long long total = 32LL * 505 * 384;
  if (idx >= total) return;
  int k = (int)(idx % 384);
  long long rb = idx / 384;
  int row = (int)(rb % 505);
  int b = (int)(rb / 505);
  float aval = 0.f, dval = 0.f;
  if (k < 360) {
    int j = k / 40, t = k - j * 40;
    const float* xb = x + (size_t)b * 2025 * 40;
    dval = xb[(size_t)(4 * row + j) * 40 + t];
    float s = 0.f;
#pragma unroll
    for (int i = 0; i < 9; ++i) s += w1[i * 9 + j] * xb[(size_t)(4 * row + i) * 40 + t];
    aval = s;
  }
  A[idx] = f2bf(aval);
  D[idx] = f2bf(dval);
}

// ---------------------------------------------------------------------------
// K3: fused cov+BN+conv1 as batched bf16 WMMA GEMM (505x505x360 per batch).
// Block tile 128(M) x 64(N), 8 waves (4x2), each wave 32x32 = 2x2 WMMA tiles.
// LDS staging uses double-buffered GLOBAL_LOAD_ASYNC_TO_LDS_B128 (ASYNCcnt):
// 3 async b128 per lane per tile; wait asynccnt<=3 overlaps next-tile fetch
// with current-tile WMMA. Row indices are clamped (uniform issue, EXEC all-1s);
// clamped rows only feed masked out-of-range outputs.
// ---------------------------------------------------------------------------
__global__ __launch_bounds__(256) void cov_conv1_gemm(const unsigned short* __restrict__ A,
                                                      const unsigned short* __restrict__ D,
                                                      const float* __restrict__ scal,
                                                      float* __restrict__ out1) {
  const int bn = blockIdx.x, bm = blockIdx.y, b = blockIdx.z;
  const int tid = threadIdx.x;
  const int lane = tid & 31, wid = tid >> 5;
  const int wave_m = wid & 3, wave_n = wid >> 2;
  const int half = lane >> 4, mrow = lane & 15;

  __shared__ __align__(16) unsigned short lA[2][128 * 32];
  __shared__ __align__(16) unsigned short lB[2][64 * 32];

  const unsigned short* Ab = A + (size_t)b * 505 * 384;
  const unsigned short* Db = D + (size_t)b * 505 * 384;

  // per-thread 16B copy slots: A gets chunks {tid, tid+256}, B gets chunk {tid}
  const int cA0 = tid, cA1 = tid + 256, cB = tid;
  int rA0 = bm * 128 + (cA0 >> 2); rA0 = rA0 > 504 ? 504 : rA0;
  int rA1 = bm * 128 + (cA1 >> 2); rA1 = rA1 > 504 ? 504 : rA1;
  int rB  = bn * 64  + (cB  >> 2); rB  = rB  > 504 ? 504 : rB;
  const int kA0 = (cA0 & 3) << 3, kA1 = (cA1 & 3) << 3, kB = (cB & 3) << 3;
  const unsigned short* gA0 = Ab + (size_t)rA0 * 384 + kA0;
  const unsigned short* gA1 = Ab + (size_t)rA1 * 384 + kA1;
  const unsigned short* gB  = Db + (size_t)rB  * 384 + kB;
  unsigned dA0[2], dA1[2], dB[2];
#pragma unroll
  for (int p = 0; p < 2; ++p) {
    dA0[p] = lds_addr_of(&lA[p][(cA0 >> 2) * 32 + kA0]);
    dA1[p] = lds_addr_of(&lA[p][(cA1 >> 2) * 32 + kA1]);
    dB[p]  = lds_addr_of(&lB[p][(cB  >> 2) * 32 + kB]);
  }

  const v8f zero = {0.f, 0.f, 0.f, 0.f, 0.f, 0.f, 0.f, 0.f};
  v8f acc[2][2];
  acc[0][0] = zero; acc[0][1] = zero; acc[1][0] = zero; acc[1][1] = zero;

  // prologue: async-fetch tile 0 into buffer 0
  async_load_b128(dA0[0], gA0);
  async_load_b128(dA1[0], gA1);
  async_load_b128(dB[0],  gB);

  for (int ks = 0; ks < 12; ++ks) {
    const int cur = ks & 1;
    if (ks > 0) __syncthreads();            // readers of buf[cur^1] (iter ks-1) done
    if (ks + 1 < 12) {                      // async-fetch next tile into other buffer
      const int nxt = cur ^ 1, k0n = (ks + 1) * 32;
      async_load_b128(dA0[nxt], gA0 + k0n);
      async_load_b128(dA1[nxt], gA1 + k0n);
      async_load_b128(dB[nxt],  gB  + k0n);
      wait_asynccnt<3>();                   // tile ks complete (in-order), tile ks+1 in flight
    } else {
      wait_asynccnt<0>();
    }
    __syncthreads();                        // all waves' tile-ks data visible

    const __bf16* pA = (const __bf16*)lA[cur];
    const __bf16* pB = (const __bf16*)lB[cur];
    union Fr { v16bf v; bf16x8 h[2]; } fa[2], fb[2];
#pragma unroll
    for (int im = 0; im < 2; ++im) {        // A frag: lane holds row m, K runs of 8
      int r = (wave_m * 32 + im * 16 + mrow) * 32;
      fa[im].h[0] = *(const bf16x8*)(pA + r + half * 8);
      fa[im].h[1] = *(const bf16x8*)(pA + r + 16 + half * 8);
    }
#pragma unroll
    for (int in = 0; in < 2; ++in) {        // B frag: lane holds col n, K runs of 8
      int r = (wave_n * 32 + in * 16 + mrow) * 32;
      fb[in].h[0] = *(const bf16x8*)(pB + r + half * 8);
      fb[in].h[1] = *(const bf16x8*)(pB + r + 16 + half * 8);
    }
#pragma unroll
    for (int im = 0; im < 2; ++im)
#pragma unroll
      for (int in = 0; in < 2; ++in)
        acc[im][in] = __builtin_amdgcn_wmma_f32_16x16x32_bf16(
            false, fa[im].v, false, fb[in].v, (short)0, acc[im][in], false, false);
  }

  const float inv = scal[0], beta = scal[1];
  float* ob = out1 + (size_t)b * 505 * 505;
#pragma unroll
  for (int im = 0; im < 2; ++im)
#pragma unroll
    for (int in = 0; in < 2; ++in)
#pragma unroll
      for (int r = 0; r < 8; ++r) {
        int rowg = bm * 128 + wave_m * 32 + im * 16 + r + half * 8;
        int colg = bn * 64 + wave_n * 32 + in * 16 + mrow;
        if (rowg < 505 && colg < 505) {
          float v = inv * acc[im][in][r] + beta;
          ob[(size_t)rowg * 505 + colg] = v > 0.f ? v : 0.f;
        }
      }
}

// ---------------------------------------------------------------------------
// K4-6: small direct convs (1 channel) with ReLU
// ---------------------------------------------------------------------------
__global__ void conv2d_kernel(const float* __restrict__ in, float* __restrict__ out,
                              const float* __restrict__ w, const float* __restrict__ bias,
                              int IN, int OUT, int KS, int STRIDE) {
  long long idx = (long long)blockIdx.x * blockDim.x + threadIdx.x;
  long long total = 32LL * OUT * OUT;
  if (idx >= total) return;
  int ox = (int)(idx % OUT);
  long long r = idx / OUT;
  int oy = (int)(r % OUT);
  int b = (int)(r / OUT);
  const float* ib = in + (size_t)b * IN * IN;
  float s = bias[0];
  for (int ky = 0; ky < KS; ++ky)
    for (int kx = 0; kx < KS; ++kx)
      s += w[ky * KS + kx] * ib[(size_t)(oy * STRIDE + ky) * IN + ox * STRIDE + kx];
  out[idx] = s > 0.f ? s : 0.f;
}

// ---------------------------------------------------------------------------
// K7/K10: out(32xN) = act( X(32xK) * W(NxK)^T + bias ) via bf16 WMMA.
// Weights converted f32->bf16 through LDS (HBM-weight-bound anyway).
// ---------------------------------------------------------------------------
__global__ __launch_bounds__(256) void fc_gemm(const float* __restrict__ X,
                                               const float* __restrict__ W,
                                               const float* __restrict__ bias,
                                               float* __restrict__ out,
                                               int K, int N, int do_relu) {
  const int n0 = blockIdx.x * 64;
  const int tid = threadIdx.x;
  const int lane = tid & 31, wid = tid >> 5;
  const int wave_m = wid & 1, wave_n = wid >> 1;
  const int half = lane >> 4, mrow = lane & 15;
  __shared__ __align__(16) unsigned short lA[32 * 32];
  __shared__ __align__(16) unsigned short lB[64 * 32];

  v8f acc = {0.f, 0.f, 0.f, 0.f, 0.f, 0.f, 0.f, 0.f};
  for (int k0 = 0; k0 < K; k0 += 32) {
    __syncthreads();
#pragma unroll
    for (int i = 0; i < 4; ++i) {                      // X tile f32 -> bf16
      int e = tid + i * 256;
      int row = e >> 5, k = e & 31;
      lA[e] = f2bf(X[(size_t)row * K + k0 + k]);
    }
#pragma unroll
    for (int i = 0; i < 8; ++i) {                      // W tile f32 -> bf16
      int e = tid + i * 256;
      int row = e >> 5, k = e & 31;
      lB[e] = f2bf(W[(size_t)(n0 + row) * K + k0 + k]);
    }
    if (k0 + 32 < K)                                   // global_prefetch_b8 next W tile
      __builtin_prefetch((const void*)&W[(size_t)(n0 + (tid & 63)) * K + k0 + 32], 0, 1);
    __syncthreads();
    const __bf16* pA = (const __bf16*)lA;
    const __bf16* pB = (const __bf16*)lB;
    union Fr { v16bf v; bf16x8 h[2]; } fa, fb;
    int ra = (wave_m * 16 + mrow) * 32;
    fa.h[0] = *(const bf16x8*)(pA + ra + half * 8);
    fa.h[1] = *(const bf16x8*)(pA + ra + 16 + half * 8);
    int rb = (wave_n * 16 + mrow) * 32;
    fb.h[0] = *(const bf16x8*)(pB + rb + half * 8);
    fb.h[1] = *(const bf16x8*)(pB + rb + 16 + half * 8);
    acc = __builtin_amdgcn_wmma_f32_16x16x32_bf16(false, fa.v, false, fb.v, (short)0,
                                                  acc, false, false);
  }
#pragma unroll
  for (int r = 0; r < 8; ++r) {
    int m = wave_m * 16 + r + half * 8;
    int n = n0 + wave_n * 16 + mrow;
    float v = acc[r] + bias[n];
    if (do_relu) v = v > 0.f ? v : 0.f;
    out[(size_t)m * N + n] = v;
  }
}

// ---------------------------------------------------------------------------
// K8: 2048-point FFT per row (Stockham radix-2, self-sorting) in LDS
// ---------------------------------------------------------------------------
__global__ __launch_bounds__(256) void fft_kernel(const float* __restrict__ in,
                                                  float* __restrict__ Xr,
                                                  float* __restrict__ Xi) {
  const int b = blockIdx.x;
  const int tid = threadIdx.x;
  __shared__ float2 bufA[2048];
  __shared__ float2 bufB[2048];
  for (int i = tid; i < 2048; i += 256) bufA[i] = make_float2(in[(size_t)b * 2048 + i], 0.f);
  __syncthreads();
  float2* src = bufA;
  float2* dst = bufB;
  int l = 1024, m = 1;
  while (l >= 1) {
    for (int u = tid; u < 1024; u += 256) {
      int j = u / m, k = u - j * m;
      float2 a = src[j * m + k];
      float2 c = src[(j + l) * m + k];
      float2 apb = make_float2(a.x + c.x, a.y + c.y);
      float2 amb = make_float2(a.x - c.x, a.y - c.y);
      float sn, cs;
      __sincosf(-PI_F * (float)j / (float)l, &sn, &cs);
      dst[2 * j * m + k] = apb;
      dst[(2 * j + 1) * m + k] = make_float2(amb.x * cs - amb.y * sn, amb.x * sn + amb.y * cs);
    }
    __syncthreads();
    float2* t = src; src = dst; dst = t;
    l >>= 1; m <<= 1;
  }
  for (int i = tid; i < 2048; i += 256) {
    Xr[(size_t)b * 2048 + i] = src[i].x;
    Xi[(size_t)b * 2048 + i] = src[i].y;
  }
}

// K8b: BN1d stats over (batch, length) for the two channels -> fold to scale/offset
__global__ void bn1d_stats_kernel(const float* __restrict__ Xr, const float* __restrict__ Xi,
                                  const float* __restrict__ g, const float* __restrict__ bb,
                                  float* __restrict__ scal) {
  __shared__ float red[256 * 4];
  int tid = threadIdx.x;
  float sr = 0.f, sr2 = 0.f, si = 0.f, si2 = 0.f;
  for (int i = tid; i < 65536; i += 256) {
    float a = Xr[i], c = Xi[i];
    sr += a; sr2 += a * a; si += c; si2 += c * c;
  }
  red[tid] = sr; red[256 + tid] = sr2; red[512 + tid] = si; red[768 + tid] = si2;
  __syncthreads();
  for (int s = 128; s > 0; s >>= 1) {
    if (tid < s) {
      red[tid] += red[tid + s]; red[256 + tid] += red[256 + tid + s];
      red[512 + tid] += red[512 + tid + s]; red[768 + tid] += red[768 + tid + s];
    }
    __syncthreads();
  }
  if (tid == 0) {
    const float N = 65536.f;
    float mr = red[0] / N,   vr = red[256] / N - mr * mr;
    float mi = red[512] / N, vi = red[768] / N - mi * mi;
    float s0 = g[0] * rsqrtf(vr + EPSV);
    float s1 = g[1] * rsqrtf(vi + EPSV);
    scal[2] = s0; scal[3] = bb[0] - mr * s0;
    scal[4] = s1; scal[5] = bb[1] - mi * s1;
  }
}

// ---------------------------------------------------------------------------
// K9: 4 stacked GRU layers, hidden=2. One wave32: lane == batch element.
// Layer 0 applies the folded BN1d to (Re,Im) at read time. Final layer ReLU'd.
// ---------------------------------------------------------------------------
__global__ void gru_kernel(const float* __restrict__ Xr, const float* __restrict__ Xi,
                           const float* __restrict__ scal,
                           const float* __restrict__ wih_all, const float* __restrict__ whh_all,
                           const float* __restrict__ bih_all, const float* __restrict__ bhh_all,
                           float* __restrict__ seqA, float* __restrict__ seqB,
                           float* __restrict__ flat) {
  const int b = threadIdx.x;
  if (b >= 32) return;
  const float sr = scal[2], orr = scal[3], si = scal[4], oi = scal[5];
  float* bufs[2] = {seqA, seqB};
  for (int l = 0; l < 4; ++l) {
    float wih[12], whh[12], bih[6], bhh[6];
    for (int i = 0; i < 12; ++i) { wih[i] = wih_all[l * 12 + i]; whh[i] = whh_all[l * 12 + i]; }
    for (int i = 0; i < 6; ++i)  { bih[i] = bih_all[l * 6 + i];  bhh[i] = bhh_all[l * 6 + i]; }
    const float* inbuf = bufs[(l + 1) & 1];
    float* outbuf = bufs[l & 1];
    float h0 = 0.f, h1 = 0.f;
    for (int t = 0; t < 2048; ++t) {
      float x0, x1;
      if (l == 0) {
        x0 = sr * Xr[(size_t)b * 2048 + t] + orr;
        x1 = si * Xi[(size_t)b * 2048 + t] + oi;
      } else {
        x0 = inbuf[((size_t)b * 2048 + t) * 2 + 0];
        x1 = inbuf[((size_t)b * 2048 + t) * 2 + 1];
      }
      float gi[6], gh[6];
#pragma unroll
      for (int gq = 0; gq < 6; ++gq) {
        gi[gq] = wih[gq * 2] * x0 + wih[gq * 2 + 1] * x1 + bih[gq];
        gh[gq] = whh[gq * 2] * h0 + whh[gq * 2 + 1] * h1 + bhh[gq];
      }
      float r0 = sigm(gi[0] + gh[0]), r1 = sigm(gi[1] + gh[1]);
      float z0 = sigm(gi[2] + gh[2]), z1 = sigm(gi[3] + gh[3]);
      float n0 = tanhf(gi[4] + r0 * gh[4]);
      float n1 = tanhf(gi[5] + r1 * gh[5]);
      h0 = (1.f - z0) * n0 + z0 * h0;
      h1 = (1.f - z1) * n1 + z1 * h1;
      outbuf[((size_t)b * 2048 + t) * 2 + 0] = h0;
      outbuf[((size_t)b * 2048 + t) * 2 + 1] = h1;
      if (l == 3) {
        flat[(size_t)b * 4096 + t * 2 + 0] = h0 > 0.f ? h0 : 0.f;
        flat[(size_t)b * 4096 + t * 2 + 1] = h1 > 0.f ? h1 : 0.f;
      }
    }
  }
}

// ---------------------------------------------------------------------------
extern "C" void kernel_launch(void* const* d_in, const int* in_sizes, int n_in,
                              void* d_out, int out_size, void* d_ws, size_t ws_size,
                              hipStream_t stream) {
  (void)in_sizes; (void)n_in; (void)out_size; (void)ws_size;
  const float* x    = (const float*)d_in[0];
  const float* w1   = (const float*)d_in[1];
  const float* cb1  = (const float*)d_in[2];
  const float* w2   = (const float*)d_in[3];
  const float* cb2  = (const float*)d_in[4];
  const float* w3   = (const float*)d_in[5];
  const float* cb3  = (const float*)d_in[6];
  const float* w4   = (const float*)d_in[7];
  const float* cb4  = (const float*)d_in[8];
  const float* fcw  = (const float*)d_in[9];
  const float* fcb  = (const float*)d_in[10];
  const float* fc1w = (const float*)d_in[11];
  const float* fc1b = (const float*)d_in[12];
  const float* gwih = (const float*)d_in[13];
  const float* gwhh = (const float*)d_in[14];
  const float* gbih = (const float*)d_in[15];
  const float* gbhh = (const float*)d_in[16];
  const float* bn2g = (const float*)d_in[17];
  const float* bn2b = (const float*)d_in[18];
  const float* bn1g = (const float*)d_in[19];
  const float* bn1b = (const float*)d_in[20];

  char* ws = (char*)d_ws;
  size_t off = 0;
  auto alloc = [&](size_t bytes) { size_t o = off; off += (bytes + 255) & ~(size_t)255; return o; };
  const size_t o_stats = alloc(512);
  const size_t packBytes = (size_t)32 * 505 * 384 * 2;        // bf16
  const size_t o_A = alloc(packBytes);
  const size_t o_D = alloc(packBytes);
  const size_t o_out1 = alloc((size_t)32 * 505 * 505 * 4);

  float* stats = (float*)(ws + o_stats);
  float* scal  = stats + 64;
  unsigned short* Abuf = (unsigned short*)(ws + o_A);
  unsigned short* Dbuf = (unsigned short*)(ws + o_D);
  float* out1 = (float*)(ws + o_out1);

  // aliases inside A region (dead after the big GEMM)
  float* Xr   = (float*)(ws + o_A);
  float* Xi   = (float*)(ws + o_A + 262144);
  float* seqA = (float*)(ws + o_A + 524288);
  float* seqB = (float*)(ws + o_A + 1048576);
  float* flat = (float*)(ws + o_A + 1572864);
  // aliases inside D region (dead after the big GEMM)
  float* c2    = (float*)(ws + o_D);
  float* c3    = (float*)(ws + o_D + 3569920);
  float* hbuf  = (float*)(ws + o_D + 4430592);
  float* fcout = (float*)(ws + o_D + 5249792);

  stats_kernel<<<32, 256, 0, stream>>>(x, stats);
  finalize1_kernel<<<1, 32, 0, stream>>>(stats, w1, cb1, bn2g, bn2b, scal);
  {
    long long total = 32LL * 505 * 384;
    pack_kernel<<<(int)((total + 255) / 256), 256, 0, stream>>>(x, w1, Abuf, Dbuf);
  }
  cov_conv1_gemm<<<dim3(8, 4, 32), 256, 0, stream>>>(Abuf, Dbuf, scal, out1);
  { long long t = 32LL * 167 * 167;
    conv2d_kernel<<<(int)((t + 255) / 256), 256, 0, stream>>>(out1, c2, w2, cb2, 505, 167, 7, 3); }
  { long long t = 32LL * 82 * 82;
    conv2d_kernel<<<(int)((t + 255) / 256), 256, 0, stream>>>(c2, c3, w3, cb3, 167, 82, 5, 2); }
  { long long t = 32LL * 80 * 80;
    conv2d_kernel<<<(int)((t + 255) / 256), 256, 0, stream>>>(c3, hbuf, w4, cb4, 82, 80, 3, 1); }
  fc_gemm<<<dim3(32), 256, 0, stream>>>(hbuf, fcw, fcb, fcout, 6400, 2048, 1);
  fft_kernel<<<32, 256, 0, stream>>>(fcout, Xr, Xi);
  bn1d_stats_kernel<<<1, 256, 0, stream>>>(Xr, Xi, bn1g, bn1b, scal);
  gru_kernel<<<1, 32, 0, stream>>>(Xr, Xi, scal, gwih, gwhh, gbih, gbhh, seqA, seqB, flat);
  fc_gemm<<<dim3(32), 256, 0, stream>>>(flat, fc1w, fc1b, (float*)d_out, 4096, 2048, 0);
}
